// TargetTokenEncoder_12043088298468
// MI455X (gfx1250) — matrix operand order repaired and checked by reading, not compiled, not run on hardware
//
#include <hip/hip_runtime.h>
#include <hip/hip_bf16.h>
#include <math.h>

// ---------------------------------------------------------------------------
// TargetTokenEncoder for MI455X (gfx1250, wave32, WMMA).
//   B=65536 tasks, S=128 labels, 10 classes, stats dim 14 (padded to 32),
//   TOKEN=256.  Both GEMMs run on v_wmma_f32_16x16x32_bf16 (f32 accumulate).
//   GELU is the branchless tanh form (error << bf16 quantization noise),
//   using hardware v_exp_f32 / v_rcp_f32 so EXEC never diverges between WMMAs.
// ---------------------------------------------------------------------------

#define S_SUP   128
#define NCLS    10
#define KPAD    32          // stats K padded 14 -> 32 (zeros)
#define TOKEN   256
#define M_TILE  64          // tasks per block
#define THREADS 256         // 8 waves

typedef __attribute__((ext_vector_type(16))) __bf16 v16bf;
typedef __attribute__((ext_vector_type(8)))  float  v8f;

union BFfrag { v16bf v; uint4 q[2]; };

__device__ __forceinline__ unsigned short f2bf(float f) {
    unsigned u = __float_as_uint(f);
    u += 0x7FFFu + ((u >> 16) & 1u);        // round-to-nearest-even
    return (unsigned short)(u >> 16);
}

// Branchless GELU (tanh form): x * sigmoid(x*(1.5957691 + 0.07135481 x^2))
//   0.5*(1+tanh(u)) == sigmoid(2u);  exp via v_exp_f32, recip via v_rcp_f32.
__device__ __forceinline__ float gelu_fast(float x) {
    float u2 = x * (1.5957691216057308f + 0.07135481627f * x * x);
    float e  = __expf(-u2);                       // v_exp_f32 path, branchless
    return x * __builtin_amdgcn_rcpf(1.0f + e);   // v_rcp_f32
}

// ---------------- weight prep: f32 -> bf16, transposed (N-major) -----------
__global__ void prep_w1(const float* __restrict__ w1, unsigned short* __restrict__ w1T) {
    int i = blockIdx.x * blockDim.x + threadIdx.x;      // TOKEN*KPAD
    if (i >= TOKEN * KPAD) return;
    int n = i >> 5, k = i & 31;                         // w1T[n][k] = w1[k][n]
    float v = (k < 14) ? w1[k * TOKEN + n] : 0.0f;
    w1T[i] = f2bf(v);
}

__global__ void prep_w2(const float* __restrict__ w2, unsigned short* __restrict__ w2T) {
    int i = blockIdx.x * blockDim.x + threadIdx.x;      // TOKEN*TOKEN
    if (i >= TOKEN * TOKEN) return;
    int n = i >> 8, k = i & 255;                        // w2T[n][k] = w2[k][n]
    w2T[i] = f2bf(w2[k * TOKEN + n]);
}

// ---------------- fused encoder -------------------------------------------
__global__ __launch_bounds__(THREADS, 1)
void encoder_kernel(const int* __restrict__ y,
                    const unsigned short* __restrict__ w1T,
                    const float* __restrict__ b1,
                    const unsigned short* __restrict__ w2T,
                    const float* __restrict__ b2,
                    float* __restrict__ out)
{
    __shared__ alignas(16) unsigned short s_stats[M_TILE][KPAD];   // 4 KB
    __shared__ alignas(16) unsigned short s_h[M_TILE][TOKEN];      // 32 KB
    __shared__ unsigned long long         s_hist[M_TILE][2];       // 1 KB

    const int tid      = threadIdx.x;
    const int taskBase = blockIdx.x * M_TILE;

    // ---- Phase 1: packed histogram (two u64 with 8-bit fields; S=128 < 255)
    if (tid < 2 * M_TILE) s_hist[tid >> 1][tid & 1] = 0ull;
    __syncthreads();
    {
        const int lt   = tid >> 2;          // local task 0..63
        const int part = tid & 3;           // 4 threads/task, 32 labels each
        const int4* p = (const int4*)(y + (size_t)(taskBase + lt) * S_SUP + part * 32);
        unsigned long long a0 = 0ull, a1 = 0ull;
        #pragma unroll
        for (int j = 0; j < 8; ++j) {
            int4 v = p[j];
            {   unsigned c = (unsigned)v.x, s = c * 8u;
                if (c < 8u) a0 += 1ull << s; else a1 += 1ull << (s - 64u); }
            {   unsigned c = (unsigned)v.y, s = c * 8u;
                if (c < 8u) a0 += 1ull << s; else a1 += 1ull << (s - 64u); }
            {   unsigned c = (unsigned)v.z, s = c * 8u;
                if (c < 8u) a0 += 1ull << s; else a1 += 1ull << (s - 64u); }
            {   unsigned c = (unsigned)v.w, s = c * 8u;
                if (c < 8u) a0 += 1ull << s; else a1 += 1ull << (s - 64u); }
        }
        atomicAdd(&s_hist[lt][0], a0);      // LDS ds_add_u64
        atomicAdd(&s_hist[lt][1], a1);
    }
    __syncthreads();

    // ---- Phase 2: stats -> bf16 rows in LDS (K padded to 32 with zeros)
    if (tid < M_TILE) {
        unsigned long long a0 = s_hist[tid][0], a1 = s_hist[tid][1];
        const float inv = 1.0f / 128.0f;    // total == S == 128 always
        float ent = 0.0f, pmax = 0.0f;
        int nnz = 0;
        #pragma unroll
        for (int c = 0; c < NCLS; ++c) {
            unsigned cnt = (c < 8) ? (unsigned)((a0 >> (8 * c)) & 0xFFull)
                                   : (unsigned)((a1 >> (8 * (c - 8))) & 0xFFull);
            float pc = (float)cnt * inv;
            ent  -= pc * __logf(pc + 1e-6f);       // hardware v_log_f32
            pmax  = fmaxf(pmax, pc);
            nnz  += (cnt > 0u) ? 1 : 0;
            s_stats[tid][c] = f2bf(pc);
        }
        s_stats[tid][10] = f2bf((float)nnz);
        s_stats[tid][11] = f2bf(ent);
        s_stats[tid][12] = f2bf((float)S_SUP);
        s_stats[tid][13] = f2bf(pmax);
        #pragma unroll
        for (int k = 14; k < KPAD; ++k) s_stats[tid][k] = 0;
    }
    __syncthreads();

    // ---- WMMA tiling: wave -> (row strip of 16, column half of 128)
    const int  lane    = tid & 31;
    const int  wave    = tid >> 5;
    const int  rowBase = (wave & 3) * 16;     // local rows 0..63
    const int  colBase = (wave >> 2) * 128;   // local cols 0..255
    const int  nIn16   = lane & 15;
    const bool hiHalf  = lane >= 16;
    const int  mRow    = rowBase + (lane & 15);   // A-fragment row (both halves)

    // ---- Phase 3: GEMM1 (K=32, one WMMA/tile) + bias + GELU -> s_h
    BFfrag aS;
    {
        // A layout: lane<16 -> K{0..15}; lane>=16 -> K{16..31} (matches B)
        const int klo = hiHalf ? 16 : 0;
        aS.q[0] = *(const uint4*)&s_stats[mRow][klo];
        aS.q[1] = *(const uint4*)&s_stats[mRow][klo + 8];
    }
    #pragma unroll
    for (int t = 0; t < 8; ++t) {
        const int ncol = colBase + t * 16 + nIn16;     // global column 0..255
        BFfrag bF;   // B layout: lane holds col N=lane&15, contiguous K run of 16
        const unsigned short* bp = w1T + ncol * KPAD + (hiHalf ? 16 : 0);
        bF.q[0] = ((const uint4*)bp)[0];
        bF.q[1] = ((const uint4*)bp)[1];
        v8f c = {0.f, 0.f, 0.f, 0.f, 0.f, 0.f, 0.f, 0.f};
        c = __builtin_amdgcn_wmma_f32_16x16x32_bf16(false, aS.v, false, bF.v,
                                                    (short)0, c, false, false);
        const float bias = b1[ncol];
        #pragma unroll
        for (int r = 0; r < 8; ++r) {
            // C/D layout: VGPR r -> row r (lanes 0-15) / r+8 (lanes 16-31)
            float x = c[r] + bias;
            float g = gelu_fast(x);
            int   m = rowBase + r + (hiHalf ? 8 : 0);
            s_h[m][colBase + t * 16 + nIn16] = f2bf(g);
        }
    }
    __syncthreads();

    // ---- Phase 4: GEMM2 (K=256 -> 8 k-steps, 8 col tiles per wave)
    v8f acc[8];
    #pragma unroll
    for (int t = 0; t < 8; ++t)
        acc[t] = (v8f){0.f, 0.f, 0.f, 0.f, 0.f, 0.f, 0.f, 0.f};

    #pragma unroll
    for (int kk = 0; kk < 8; ++kk) {
        BFfrag aH;
        const int klo = kk * 32 + (hiHalf ? 16 : 0);
        aH.q[0] = *(const uint4*)&s_h[mRow][klo];
        aH.q[1] = *(const uint4*)&s_h[mRow][klo + 8];
        #pragma unroll
        for (int t = 0; t < 8; ++t) {
            const int ncol = colBase + t * 16 + nIn16;
            const unsigned short* bp = w2T + ncol * TOKEN + kk * 32 + (hiHalf ? 16 : 0);
            BFfrag bF;
            bF.q[0] = ((const uint4*)bp)[0];
            bF.q[1] = ((const uint4*)bp)[1];
            acc[t] = __builtin_amdgcn_wmma_f32_16x16x32_bf16(false, aH.v, false, bF.v,
                                                             (short)0, acc[t], false, false);
        }
    }

    // ---- Phase 5: bias + store f32 output
    #pragma unroll
    for (int t = 0; t < 8; ++t) {
        const int   ncol = colBase + t * 16 + nIn16;
        const float bias = b2[ncol];
        #pragma unroll
        for (int r = 0; r < 8; ++r) {
            const int m = rowBase + r + (hiHalf ? 8 : 0);
            out[(size_t)(taskBase + m) * TOKEN + ncol] = acc[t][r] + bias;
        }
    }
}

// ---------------------------------------------------------------------------
extern "C" void kernel_launch(void* const* d_in, const int* in_sizes, int n_in,
                              void* d_out, int out_size, void* d_ws, size_t ws_size,
                              hipStream_t stream) {
    const int*   y  = (const int*)d_in[0];      // [B,128] int32
    const float* w1 = (const float*)d_in[1];    // [14,256]
    const float* b1 = (const float*)d_in[2];    // [256]
    const float* w2 = (const float*)d_in[3];    // [256,256]
    const float* b2 = (const float*)d_in[4];    // [256]
    float*       out = (float*)d_out;           // [B,256] f32

    unsigned short* w1T = (unsigned short*)d_ws;           // 256*32  bf16 = 16 KB
    unsigned short* w2T = w1T + TOKEN * KPAD;              // 256*256 bf16 = 128 KB

    prep_w1<<<(TOKEN * KPAD  + 255) / 256, 256, 0, stream>>>(w1, w1T);
    prep_w2<<<(TOKEN * TOKEN + 255) / 256, 256, 0, stream>>>(w2, w2T);

    const int nTasks = in_sizes[0] / S_SUP;                // 65536
    const int nBlocks = (nTasks + M_TILE - 1) / M_TILE;    // 1024
    encoder_kernel<<<nBlocks, THREADS, 0, stream>>>(y, w1T, b1, w2T, b2, out);
}
